// MixtureOfExperts_29678224015411
// MI455X (gfx1250) — matrix operand order repaired and checked by reading, not compiled, not run on hardware
//
#include <hip/hip_runtime.h>
#include <hip/hip_bf16.h>

// Problem constants (from reference): B=4, S=2048, D=512, H=1024, O=512, E=8, TOPK=2
#define T_TOKENS 8192
#define D_IN     512
#define H_DIM    1024
#define O_DIM    512
#define E_NUM    8

typedef __attribute__((ext_vector_type(16))) __bf16 v16bf;
typedef __attribute__((ext_vector_type(8)))  float  v8f;

#if __has_builtin(__builtin_amdgcn_tensor_load_to_lds) && \
    __has_builtin(__builtin_amdgcn_s_wait_tensorcnt)
#define USE_TDM 1
typedef __attribute__((ext_vector_type(4))) unsigned int u32x4;
typedef __attribute__((ext_vector_type(8))) int          i32x8;
typedef __attribute__((ext_vector_type(4))) int          i32x4;
#else
#define USE_TDM 0
#endif

// ---------------------------------------------------------------- helpers
__device__ __forceinline__ unsigned short f2bf(float f) {
    unsigned int u = __float_as_uint(f);
    u += 0x7FFFu + ((u >> 16) & 1u);          // round-to-nearest-even
    return (unsigned short)(u >> 16);
}

__device__ __forceinline__ float gelu_f(float v) {
#if __has_builtin(__builtin_amdgcn_tanhf)
    // CDNA5 native V_TANH_F32 — branch-free
    float c = 0.7978845608028654f * (v + 0.044715f * v * v * v);
    return 0.5f * v * (1.0f + __builtin_amdgcn_tanhf(c));
#else
    // branch-free sigmoid approximation (v_exp_f32 fast path)
    return v * (1.0f / (1.0f + __expf(-1.702f * v)));
#endif
}

// CDNA5 async global->LDS copy (ASYNCcnt-tracked), 16 bytes per lane.
__device__ __forceinline__ void async_b128(unsigned lds_byte_addr, const void* gptr) {
    asm volatile("global_load_async_to_lds_b128 %0, %1, off"
                 :: "v"(lds_byte_addr), "v"((unsigned long long)gptr)
                 : "memory");
}
__device__ __forceinline__ void wait_async0(void) {
    asm volatile("s_wait_asynccnt 0" ::: "memory");
}

// A fragment (16x32 bf16, row-major LDS [m][k], stride in elements).
__device__ __forceinline__ v16bf load_frag_a(const unsigned short* base, int row,
                                             int stride, int k0, int lane) {
    int r = row + (lane & 15);
    int k = k0 + ((lane >> 4) << 3);
    const unsigned short* p = base + r * stride + k;
    union { uint4 q[2]; v16bf v; } u;
    u.q[0] = *(const uint4*)(p);
    u.q[1] = *(const uint4*)(p + 16);
    return u.v;
}

// B fragment (32x16 bf16, column-major LDS [n][k], stride in elements).
__device__ __forceinline__ v16bf load_frag_b(const unsigned short* base, int col,
                                             int stride, int k0, int lane) {
    int c = col + (lane & 15);
    int k = k0 + ((lane >> 4) << 3);
    const unsigned short* p = base + c * stride + k;
    union { uint4 q[2]; v16bf v; } u;
    u.q[0] = *(const uint4*)(p);
    u.q[1] = *(const uint4*)(p + 16);
    return u.v;
}

// ---------------------------------------------------------------- kernel 0: zero out
__global__ void zero_f32(float* __restrict__ p, int n) {
    int i = blockIdx.x * blockDim.x + threadIdx.x;
    if (i < n) p[i] = 0.0f;
}

// ---------------------------------------------------------------- prep: bf16 conversions
__global__ void prep_x(const float* __restrict__ x, unsigned short* __restrict__ xbf,
                       int n) {
    int i = blockIdx.x * blockDim.x + threadIdx.x;
    if (i < n) xbf[i] = f2bf(x[i]);
}

// W1 [E][D][H] f32 -> W1t [E][H][D] bf16 (transposed so K is contiguous per N row)
__global__ void prep_w1(const float* __restrict__ W1, unsigned short* __restrict__ W1t) {
    size_t i = (size_t)blockIdx.x * 256 + threadIdx.x;
    if (i >= (size_t)E_NUM * H_DIM * D_IN) return;
    int    k  = (int)(i % D_IN);
    size_t nh = i / D_IN;
    int    n  = (int)(nh % H_DIM);
    int    e  = (int)(nh / H_DIM);
    W1t[i] = f2bf(W1[((size_t)e * D_IN + k) * H_DIM + n]);
}

// W2 [E][H][O] f32 -> W2t [E][O][H] bf16
__global__ void prep_w2(const float* __restrict__ W2, unsigned short* __restrict__ W2t) {
    size_t i = (size_t)blockIdx.x * 256 + threadIdx.x;
    if (i >= (size_t)E_NUM * O_DIM * H_DIM) return;
    int    k  = (int)(i % H_DIM);
    size_t no = i / H_DIM;
    int    n  = (int)(no % O_DIM);
    int    e  = (int)(no / O_DIM);
    W2t[i] = f2bf(W2[((size_t)e * H_DIM + k) * O_DIM + n]);
}

// ---------------------------------------------------------------- kernel 1: router
__global__ void __launch_bounds__(256)
moe_router(const float* __restrict__ x, const float* __restrict__ Wg,
           float* __restrict__ gates, int* __restrict__ topi) {
    __shared__ float wg[D_IN * E_NUM];                 // 16 KB
    for (int i = threadIdx.x; i < D_IN * E_NUM; i += 256) wg[i] = Wg[i];
    __syncthreads();

    int t = blockIdx.x * 256 + threadIdx.x;
    if (t >= T_TOKENS) return;

    const float* xr = x + (size_t)t * D_IN;
    float acc[E_NUM];
#pragma unroll
    for (int e = 0; e < E_NUM; ++e) acc[e] = 0.0f;
    for (int d = 0; d < D_IN; ++d) {
        float xv = xr[d];
#pragma unroll
        for (int e = 0; e < E_NUM; ++e) acc[e] += xv * wg[d * E_NUM + e];
    }
    float mx = acc[0];
#pragma unroll
    for (int e = 1; e < E_NUM; ++e) mx = fmaxf(mx, acc[e]);
    float s = 0.0f;
#pragma unroll
    for (int e = 0; e < E_NUM; ++e) { acc[e] = __expf(acc[e] - mx); s += acc[e]; }
    float inv = 1.0f / s;
#pragma unroll
    for (int e = 0; e < E_NUM; ++e) acc[e] *= inv;
    int i1 = 0; float p1 = acc[0];
#pragma unroll
    for (int e = 1; e < E_NUM; ++e) if (acc[e] > p1) { p1 = acc[e]; i1 = e; }
    int i2 = -1; float p2 = -1.0f;
#pragma unroll
    for (int e = 0; e < E_NUM; ++e) if (e != i1 && acc[e] > p2) { p2 = acc[e]; i2 = e; }
#pragma unroll
    for (int e = 0; e < E_NUM; ++e)
        gates[(size_t)t * E_NUM + e] = (e == i1) ? p1 : ((e == i2) ? p2 : 0.0f);
    topi[2 * t]     = i1;
    topi[2 * t + 1] = i2;
}

// ---------------------------------------------------------------- kernel 2: compaction
__global__ void moe_compact(const int* __restrict__ topi, int* __restrict__ lists,
                            int* __restrict__ counts) {
    int e = threadIdx.x;
    if (e >= E_NUM) return;
    int c = 0;
    for (int t = 0; t < T_TOKENS; ++t) {
        int i1 = topi[2 * t], i2 = topi[2 * t + 1];
        if (i1 == e || i2 == e) lists[e * T_TOKENS + c++] = t;
    }
    counts[e] = c;
}

// ---------------------------------------------------------------- kernel 3: fused expert MLP
// Block = (32 gathered tokens) x (1 expert). 256 threads = 8 waves.
// LDS layout (dynamic, 234496 B); ALL fragment reads go through ONE lds base
// pointer + integer offsets so addrspace inference keeps them ds_load_b128.
//   hs  @ elem 0      : bf16 [32][1024+8]  (layer-1 activations, A of layer 2)
//   xs  @ elem 33024  : bf16 [32][512+8]   (gathered x tile, A of layer 1)
//   wb  @ elem 49664  : 2 x bf16 [128][256+8] ping-pong K-half W panels
#define SMEM_BYTES   234496
#define XS_STRIDE    520
#define HS_STRIDE    1032
#define WS_STRIDE    264
#define WKH          256                 // K-half depth staged per stage
#define HS_OFF_E     0
#define XS_OFF_E     33024
#define WB_OFF_E     49664
#define WB_STRIDE_E  33792
#define XS_OFF_B     (XS_OFF_E * 2)
#define WB_OFF_B     (WB_OFF_E * 2)
#define WB_STRIDE_B  (WB_STRIDE_E * 2)

#if USE_TDM
// One Tensor-Data-Mover descriptor moves the whole 128x256-bf16 K-half panel,
// inserting 16B of LDS padding after every 512B row (pad_interval=6: 128 DWORDs,
// pad_amount=3: 4 DWORDs) -> reproduces the [128][264] conflict-free layout.
// Issued by wave 0 only; tracked with TENSORcnt.
__device__ __forceinline__ void issue_w_stage_tdm(unsigned buf_byte,
                                                  const unsigned short* Wt_e,
                                                  int Kfull, int n0, int k0) {
    unsigned long long ga =
        (unsigned long long)(const void*)(Wt_e + (size_t)n0 * Kfull + k0);
    u32x4 g0 = {};
    g0[0] = 1u;                                            // count=1 (valid user D#)
    g0[1] = buf_byte;                                      // lds_addr
    g0[2] = (unsigned)ga;                                  // global_addr[31:0]
    g0[3] = (unsigned)((ga >> 32) & 0x01FFFFFFu)           // global_addr[56:32]
            | 0x80000000u;                                 // type=2 ("image")
    i32x8 g1 = {};
    g1[0] = (int)((1u << 16) |                             // data_size = 2 bytes
                  (1u << 20) |                             // pad_enable
                  (6u << 22) |                             // pad_interval: 128 DW
                  (3u << 25));                             // pad_amount: 4 DW
    g1[1] = (int)(((unsigned)Kfull & 0xFFFFu) << 16);      // tensor_dim0[15:0]
    g1[2] = (int)((((unsigned)Kfull >> 16) & 0xFFFFu) |    // tensor_dim0[31:16]
                  (128u << 16));                           // tensor_dim1[15:0] = 128
    g1[3] = (int)(((unsigned)WKH & 0xFFFFu) << 16);        // tile_dim0 = 256
    g1[4] = 128;                                           // tile_dim1 = 128
    g1[5] = Kfull;                                         // tensor_dim0_stride[31:0]
    i32x4 z4 = {};
    i32x8 z8 = {};
    // 6-arg variant (this toolchain): (g0, g1, g2, g3, g4, cpol)
    __builtin_amdgcn_tensor_load_to_lds(g0, g1, z4, z4, z8, 0);
}
#endif

// Fallback: per-lane ASYNCcnt copies (16B each), 16 per thread.
__device__ __forceinline__ void issue_w_stage(unsigned buf_byte, const unsigned short* Wt_e,
                                              int Kfull, int n0, int k0, int tid) {
    for (int idx = tid; idx < 128 * (WKH / 8); idx += 256) {
        int nl = idx >> 5, c8 = idx & 31;
        async_b128(buf_byte + (unsigned)(nl * WS_STRIDE + c8 * 8) * 2,
                   Wt_e + (size_t)(n0 + nl) * Kfull + k0 + c8 * 8);
    }
}

__global__ void __launch_bounds__(256)
moe_expert(const unsigned short* __restrict__ xbf,
           const unsigned short* __restrict__ W1t, const float* __restrict__ b1g,
           const unsigned short* __restrict__ W2t, const float* __restrict__ b2g,
           const float* __restrict__ gates, const int* __restrict__ lists,
           const int* __restrict__ counts, float* __restrict__ out) {
    const int e     = blockIdx.y;
    const int cnt   = counts[e];
    const int tile0 = blockIdx.x * 32;
    if (tile0 >= cnt) return;                       // uniform early exit

    extern __shared__ char smem_raw[];
    unsigned short* lds = (unsigned short*)smem_raw;          // single LDS base
    const unsigned  lds_base = (unsigned)(size_t)(void*)smem_raw;
    __shared__ int   tok[32];
    __shared__ float gts[32];

    const int tid  = threadIdx.x;
    const int lane = tid & 31;
    const int w    = tid >> 5;
    const int mi   = w & 1;        // M tile (2 x 16 rows)
    const int nq   = w >> 1;       // N quad (4 x 32 cols per 128-wide panel)

    if (tid < 32) {
        int t = 0; float g = 0.0f;
        if (tile0 + tid < cnt) {
            t = lists[e * T_TOKENS + tile0 + tid];
            g = gates[(size_t)t * E_NUM + e];
        }
        tok[tid] = t; gts[tid] = g;                 // padded rows: token 0, gate 0
    }
    __syncthreads();

    const unsigned short* W1e = W1t + (size_t)e * H_DIM * D_IN;   // [H][D]
    const unsigned short* W2e = W2t + (size_t)e * O_DIM * H_DIM;  // [O][H]

// stage-issue / stage-wait selection
#if USE_TDM
#define ISSUE_W(buf, We, Kf, n0, k0)                                        \
    do { if (w == 0) issue_w_stage_tdm((buf), (We), (Kf), (n0), (k0)); } while (0)
#define WAIT_W()                                                            \
    do { if (w == 0) __builtin_amdgcn_s_wait_tensorcnt(0); } while (0)
#else
#define ISSUE_W(buf, We, Kf, n0, k0) issue_w_stage((buf), (We), (Kf), (n0), (k0), tid)
#define WAIT_W() wait_async0()
#endif

    // ---- prologue: async gather x tile + first layer-1 W stage into buf 0 ----
    for (int idx = tid; idx < 32 * (D_IN / 8); idx += 256) {      // 8 per thread
        int r = idx >> 6, c8 = idx & 63;
        async_b128(lds_base + XS_OFF_B + (unsigned)(r * XS_STRIDE + c8 * 8) * 2,
                   xbf + (size_t)tok[r] * D_IN + c8 * 8);
    }
    ISSUE_W(lds_base + WB_OFF_B, W1e, D_IN, 0, 0);
    wait_async0();
    WAIT_W();
    __syncthreads();

    int pb = 0;                                    // current (ready) buffer parity

    // ---------------- layer 1: 8 N-panels x 2 K-halves ------------------------
    for (int np = 0; np < 8; ++np) {
        v8f a0 = {}; v8f a1 = {};
        for (int kh = 0; kh < 2; ++kh) {
            int sn = np * 2 + kh + 1;              // next stage index
            unsigned nb = lds_base + WB_OFF_B + (unsigned)(pb ^ 1) * WB_STRIDE_B;
            if (sn < 16)
                ISSUE_W(nb, W1e, D_IN, (sn >> 1) * 128, (sn & 1) * WKH);
            else                                    // prefetch layer-2 stage 0
                ISSUE_W(nb, W2e, H_DIM, 0, 0);

            const unsigned short* wc = lds + WB_OFF_E + (unsigned)pb * WB_STRIDE_E;
#pragma unroll
            for (int kk = 0; kk < WKH; kk += 32) {
                v16bf af  = load_frag_a(lds + XS_OFF_E, mi * 16, XS_STRIDE,
                                        kh * WKH + kk, lane);
                v16bf bf0 = load_frag_b(wc, nq * 32,      WS_STRIDE, kk, lane);
                v16bf bf1 = load_frag_b(wc, nq * 32 + 16, WS_STRIDE, kk, lane);
                a0 = __builtin_amdgcn_wmma_f32_16x16x32_bf16(false, af, false, bf0,
                                                             (short)0, a0, false, false);
                a1 = __builtin_amdgcn_wmma_f32_16x16x32_bf16(false, af, false, bf1,
                                                             (short)0, a1, false, false);
            }
            if (kh == 1) {                         // bias + gelu -> hs (bf16)
                int m0  = mi * 16 + ((lane >> 4) << 3);
                int nc0 = np * 128 + nq * 32 + (lane & 15);
                float bb0 = b1g[e * H_DIM + nc0];
                float bb1 = b1g[e * H_DIM + nc0 + 16];
#pragma unroll
                for (int r = 0; r < 8; ++r) {
                    lds[HS_OFF_E + (m0 + r) * HS_STRIDE + nc0]      =
                        f2bf(gelu_f(a0[r] + bb0));
                    lds[HS_OFF_E + (m0 + r) * HS_STRIDE + nc0 + 16] =
                        f2bf(gelu_f(a1[r] + bb1));
                }
            }
            WAIT_W();
            __syncthreads();
            pb ^= 1;
        }
    }

    // ---------------- layer 2: 4 N-panels x 4 K-halves ------------------------
    for (int np2 = 0; np2 < 4; ++np2) {
        v8f a0 = {}; v8f a1 = {};
        for (int kh = 0; kh < 4; ++kh) {
            int sn = np2 * 4 + kh + 1;
            if (sn < 16) {
                unsigned nb = lds_base + WB_OFF_B + (unsigned)(pb ^ 1) * WB_STRIDE_B;
                ISSUE_W(nb, W2e, H_DIM, (sn >> 2) * 128, (sn & 3) * WKH);
            }
            const unsigned short* wc = lds + WB_OFF_E + (unsigned)pb * WB_STRIDE_E;
#pragma unroll
            for (int kk = 0; kk < WKH; kk += 32) {
                v16bf af  = load_frag_a(lds + HS_OFF_E, mi * 16, HS_STRIDE,
                                        kh * WKH + kk, lane);
                v16bf bf0 = load_frag_b(wc, nq * 32,      WS_STRIDE, kk, lane);
                v16bf bf1 = load_frag_b(wc, nq * 32 + 16, WS_STRIDE, kk, lane);
                a0 = __builtin_amdgcn_wmma_f32_16x16x32_bf16(false, af, false, bf0,
                                                             (short)0, a0, false, false);
                a1 = __builtin_amdgcn_wmma_f32_16x16x32_bf16(false, af, false, bf1,
                                                             (short)0, a1, false, false);
            }
            if (kh == 3) {                         // bias + gate-weighted scatter
                int m0 = mi * 16 + ((lane >> 4) << 3);
                int nc = np2 * 128 + nq * 32 + (lane & 15);
                float bb0 = b2g[e * O_DIM + nc];
                float bb1 = b2g[e * O_DIM + nc + 16];
#pragma unroll
                for (int r = 0; r < 8; ++r) {
                    int m = m0 + r;
                    float g = gts[m];
                    atomicAdd(out + (size_t)tok[m] * O_DIM + nc,      (a0[r] + bb0) * g);
                    atomicAdd(out + (size_t)tok[m] * O_DIM + nc + 16, (a1[r] + bb1) * g);
                }
            }
            WAIT_W();
            __syncthreads();
            pb ^= 1;
        }
    }
#undef ISSUE_W
#undef WAIT_W
}

// ---------------------------------------------------------------- launch
extern "C" void kernel_launch(void* const* d_in, const int* in_sizes, int n_in,
                              void* d_out, int out_size, void* d_ws, size_t ws_size,
                              hipStream_t stream) {
    (void)in_sizes; (void)n_in; (void)ws_size;
    const float* x  = (const float*)d_in[0];
    const float* Wg = (const float*)d_in[1];
    const float* W1 = (const float*)d_in[2];
    const float* b1 = (const float*)d_in[3];
    const float* W2 = (const float*)d_in[4];
    const float* b2 = (const float*)d_in[5];
    float* out = (float*)d_out;

    // workspace layout (bytes)
    char* ws = (char*)d_ws;
    unsigned short* xbf = (unsigned short*)(ws);                       // T*D bf16    = 8 MiB
    unsigned short* W1t = (unsigned short*)(ws + 8388608);             // E*H*D bf16  = 8 MiB
    unsigned short* W2t = (unsigned short*)(ws + 16777216);            // E*O*H bf16  = 8 MiB
    float* gates  = (float*)(ws + 25165824);                           // T*8 f32
    int*   topi   = (int*)  (ws + 25427968);                           // T*2 i32
    int*   lists  = (int*)  (ws + 25493504);                           // E*T i32
    int*   counts = (int*)  (ws + 25755648);                           // E i32

    (void)hipFuncSetAttribute((const void*)moe_expert,
                              hipFuncAttributeMaxDynamicSharedMemorySize, SMEM_BYTES);

    int n_out = out_size;                                              // T * O
    zero_f32<<<(n_out + 255) / 256, 256, 0, stream>>>(out, n_out);

    int nx = T_TOKENS * D_IN;
    prep_x<<<(nx + 255) / 256, 256, 0, stream>>>(x, xbf, nx);
    prep_w1<<<(E_NUM * H_DIM * D_IN + 255) / 256, 256, 0, stream>>>(W1, W1t);
    prep_w2<<<(E_NUM * O_DIM * H_DIM + 255) / 256, 256, 0, stream>>>(W2, W2t);

    moe_router<<<T_TOKENS / 256, 256, 0, stream>>>(x, Wg, gates, topi);
    moe_compact<<<1, 32, 0, stream>>>(topi, lists, counts);

    dim3 grid(T_TOKENS / 32, E_NUM);
    moe_expert<<<grid, 256, SMEM_BYTES, stream>>>(xbf, W1t, b1, W2t, b2,
                                                  gates, lists, counts, out);
}
